// Block_34668976014014
// MI455X (gfx1250) — matrix operand order repaired and checked by reading, not compiled, not run on hardware
//
#include <hip/hip_runtime.h>

// ---------------------------------------------------------------------------
// Transformer block forward for MI455X (gfx1250, wave32, WMMA).
// B=2, T=2048, C=1024, H=16, HS=64.  All matmuls in bf16 WMMA w/ f32 accum.
// ---------------------------------------------------------------------------

static constexpr int BB   = 2;
static constexpr int TT   = 2048;
static constexpr int CC   = 1024;
static constexpr int HH   = 16;
static constexpr int HSZ  = 64;
static constexpr int CF   = 4096;          // 4*C
static constexpr int MTOT = BB * TT;       // 4096 rows
static constexpr float EPS = 1e-5f;
static constexpr float ATT_SCALE = 0.03125f;  // C^-0.5 = 1/32

typedef __attribute__((ext_vector_type(16))) __bf16 v16bf;
typedef __attribute__((ext_vector_type(8)))  float  v8f;

union BF16x16 {           // one WMMA A/B fragment (16 bf16 = 8 VGPRs)
    uint4 u[2];
    v16bf v;
};

__device__ __forceinline__ unsigned short f2bf(float f) {
    union { float f; unsigned u; } a; a.f = f;
    unsigned r = a.u + 0x7FFFu + ((a.u >> 16) & 1u);   // round-to-nearest-even
    return (unsigned short)(r >> 16);
}

__device__ __forceinline__ float rowmax16(float v) {
    #pragma unroll
    for (int o = 8; o; o >>= 1) v = fmaxf(v, __shfl_xor(v, o, 16));
    return v;
}
__device__ __forceinline__ float rowsum16(float v) {
    #pragma unroll
    for (int o = 8; o; o >>= 1) v += __shfl_xor(v, o, 16);
    return v;
}

// Async global -> LDS copy of 16 bytes per lane (ASYNCcnt-tracked DMA).
// LDS offset = low 32 bits of the flat shared pointer (aperture | wg offset).
__device__ __forceinline__ void async_copy_b128(const void* g, void* lds) {
    unsigned loff = (unsigned)(uintptr_t)lds;
    asm volatile("global_load_async_to_lds_b128 %0, %1, off"
                 :: "v"(loff), "v"(g) : "memory");
}
__device__ __forceinline__ void wait_async0() {
    asm volatile("s_wait_asynccnt 0x0" ::: "memory");
}

// ---------------------------------------------------------------------------
// Weight transpose + fp32 -> bf16:  W[K,N] -> Wt[N,K]
// ---------------------------------------------------------------------------
__global__ __launch_bounds__(256)
void transpose_bf16(const float* __restrict__ W, unsigned short* __restrict__ Wt,
                    int K, int N) {
    __shared__ float tile[32][33];
    int n0 = blockIdx.x * 32, k0 = blockIdx.y * 32;
    int tx = threadIdx.x, ty = threadIdx.y;   // (32, 8)
    #pragma unroll
    for (int i = 0; i < 4; i++)
        tile[ty + i * 8][tx] = W[(size_t)(k0 + ty + i * 8) * N + n0 + tx];
    __syncthreads();
    #pragma unroll
    for (int i = 0; i < 4; i++)
        Wt[(size_t)(n0 + ty + i * 8) * K + k0 + tx] = f2bf(tile[tx][ty + i * 8]);
}

// ---------------------------------------------------------------------------
// LayerNorm(row of C=1024 fp32) -> bf16
// ---------------------------------------------------------------------------
__global__ __launch_bounds__(256)
void ln_bf16(const float* __restrict__ x, const float* __restrict__ g,
             const float* __restrict__ bta, unsigned short* __restrict__ out) {
    int row = blockIdx.x;
    const float* xr = x + (size_t)row * CC;
    float v[4], s = 0.f, s2 = 0.f;
    #pragma unroll
    for (int i = 0; i < 4; i++) {
        v[i] = xr[threadIdx.x + i * 256];
        s += v[i]; s2 += v[i] * v[i];
    }
    __shared__ float red0[256], red1[256];
    red0[threadIdx.x] = s; red1[threadIdx.x] = s2;
    __syncthreads();
    for (int o = 128; o; o >>= 1) {
        if (threadIdx.x < o) {
            red0[threadIdx.x] += red0[threadIdx.x + o];
            red1[threadIdx.x] += red1[threadIdx.x + o];
        }
        __syncthreads();
    }
    float mu   = red0[0] * (1.f / CC);
    float var  = red1[0] * (1.f / CC) - mu * mu;
    float rstd = rsqrtf(var + EPS);
    #pragma unroll
    for (int i = 0; i < 4; i++) {
        int c = threadIdx.x + i * 256;
        out[(size_t)row * CC + c] = f2bf((v[i] - mu) * rstd * g[c] + bta[c]);
    }
}

// ---------------------------------------------------------------------------
// WMMA GEMM:  out[M,N] = A[M,K](bf16) * Bt[N,K]^T(bf16)  (+epilogue)
//   MODE 0: store bf16
//   MODE 1: store bf16 transposed per-batch -> vT[b, n, t]   (for attention V)
//   MODE 2: f32 out = resid + acc + bias
//   MODE 3: bf16 out = relu(acc + bias)
// Block: 256 thr = 8 waves; block tile 128x128; wave tile 32x64 (2x4 tiles).
// B slab (128 x 32 K) staged via ASYNC global->LDS DMA, double-buffered so the
// DMA of slab i+1 overlaps the WMMAs on slab i.  A fragments direct-from-global
// with register double-buffering.
// ---------------------------------------------------------------------------
template <int MODE>
__global__ __launch_bounds__(256)
void gemm_wmma(const unsigned short* __restrict__ A,
               const unsigned short* __restrict__ Bt,
               const float* __restrict__ bias,
               const float* __restrict__ resid,
               float* __restrict__ outF,
               unsigned short* __restrict__ outH,
               int M, int N, int K) {
    __shared__ __align__(16) unsigned short bslab[2][128 * 32];   // 2 x 8KB

    const int lane = threadIdx.x & 31;
    const int wave = threadIdx.x >> 5;
    const int wm = wave >> 1, wn = wave & 1;
    const int n0b = blockIdx.x * 128;
    const int m0 = blockIdx.y * 128 + wm * 32;
    const int n0 = n0b + wn * 64;
    const int half = lane >> 4, lr = lane & 15;

    // Per-lane A base pointers (K advances by +32 elements each step).
    const unsigned short* pa0 = A + (size_t)(m0 + lr)      * K + half * 8;
    const unsigned short* pa1 = A + (size_t)(m0 + 16 + lr) * K + half * 8;

    // Async B-slab staging: 8KB per slab, 256 threads x 2 chunks x 16B.
    const int tid = threadIdx.x;
    const int srow = tid >> 2;            // 0..63
    const int skk  = (tid & 3) * 8;       // k-element offset of 16B chunk
    auto issue_slab = [&](int k0, int s) {
        #pragma unroll
        for (int cnk = 0; cnk < 2; cnk++) {
            int row = cnk * 64 + srow;    // 0..127
            async_copy_b128(Bt + (size_t)(n0b + row) * K + k0 + skk,
                            &bslab[s][row * 32 + skk]);
        }
    };

    v8f c[2][4] = {};
    BF16x16 a[2][2];
    auto load_a = [&](int k0, int s) {
        a[s][0].u[0] = *(const uint4*)(pa0 + k0);
        a[s][0].u[1] = *(const uint4*)(pa0 + k0 + 16);
        a[s][1].u[0] = *(const uint4*)(pa1 + k0);
        a[s][1].u[1] = *(const uint4*)(pa1 + k0 + 16);
    };

    issue_slab(0, 0);
    load_a(0, 0);
    int buf = 0;
    for (int k0 = 0; k0 < K; k0 += 32) {
        wait_async0();        // this wave's portion of slab `buf` has landed
        __syncthreads();      // all portions visible; other buffer free to reuse
        if (k0 + 32 < K) {    // overlap next slab DMA + next A loads with WMMAs
            issue_slab(k0 + 32, buf ^ 1);
            load_a(k0 + 32, buf ^ 1);
        }
        // B fragments from LDS (ds_load_b128 x2 per fragment).
        BF16x16 bfr[4];
        #pragma unroll
        for (int tj = 0; tj < 4; tj++) {
            const unsigned short* p =
                &bslab[buf][(wn * 64 + tj * 16 + lr) * 32 + half * 8];
            bfr[tj].u[0] = *(const uint4*)(p);
            bfr[tj].u[1] = *(const uint4*)(p + 16);
        }
        #pragma unroll
        for (int ti = 0; ti < 2; ti++)
            #pragma unroll
            for (int tj = 0; tj < 4; tj++)
                c[ti][tj] = __builtin_amdgcn_wmma_f32_16x16x32_bf16(
                    false, a[buf][ti].v, false, bfr[tj].v, (short)0,
                    c[ti][tj], false, false);
        buf ^= 1;
    }

    #pragma unroll
    for (int ti = 0; ti < 2; ti++)
        #pragma unroll
        for (int tj = 0; tj < 4; tj++)
            #pragma unroll
            for (int r = 0; r < 8; r++) {
                int row = m0 + ti * 16 + r + half * 8;   // D layout
                int col = n0 + tj * 16 + lr;
                float val = c[ti][tj][r];
                if constexpr (MODE == 0) {
                    outH[(size_t)row * N + col] = f2bf(val);
                } else if constexpr (MODE == 1) {
                    int bb = row / TT, t = row % TT;
                    outH[((size_t)bb * CC + col) * TT + t] = f2bf(val);
                } else if constexpr (MODE == 2) {
                    outF[(size_t)row * N + col] =
                        resid[(size_t)row * N + col] + val + bias[col];
                } else {
                    float z = val + bias[col];
                    outH[(size_t)row * N + col] = f2bf(z > 0.f ? z : 0.f);
                }
            }
}

// ---------------------------------------------------------------------------
// Flash attention, causal. One wave (32 thr) per (b, h, 16-query tile).
// q,k: bf16 [B,T,C] (head-major cols).  vT: bf16 [B, C, T].  out: bf16 [B,T,C]
// ---------------------------------------------------------------------------
__global__ __launch_bounds__(32)
void attn_flash(const unsigned short* __restrict__ q,
                const unsigned short* __restrict__ k,
                const unsigned short* __restrict__ vT,
                unsigned short* __restrict__ out) {
    __shared__ __align__(16) unsigned short pLds[16 * 32];

    const int lane = threadIdx.x;
    const int half = lane >> 4, lr = lane & 15;
    const int wid = blockIdx.x;
    const int qt = wid % (TT / 16);
    const int h  = (wid / (TT / 16)) % HH;
    const int b  = wid / ((TT / 16) * HH);
    const int q0 = qt * 16;

    BF16x16 aq[2];
    #pragma unroll
    for (int dk = 0; dk < 2; dk++) {
        const unsigned short* p =
            q + ((size_t)(b * TT + q0 + lr)) * CC + h * HSZ + dk * 32 + half * 8;
        aq[dk].u[0] = *(const uint4*)(p);
        aq[dk].u[1] = *(const uint4*)(p + 16);
    }

    const unsigned short* kbase =
        k + ((size_t)(b * TT + lr)) * CC + h * HSZ + half * 8;
    const unsigned short* vbase =
        vT + ((size_t)((b * HH + h) * HSZ + lr)) * TT + half * 8;

    v8f acc[4] = {};
    float m[8], l[8];
    #pragma unroll
    for (int r = 0; r < 8; r++) { m[r] = -1e30f; l[r] = 0.f; }

    for (int kc = 0; kc <= q0; kc += 32) {
        // ---- issue ALL chunk loads first: K frags (t,dk) and V frags (dt) ----
        BF16x16 bk[2][2], bv[4];
        #pragma unroll
        for (int t = 0; t < 2; t++)
            #pragma unroll
            for (int dk = 0; dk < 2; dk++) {
                const unsigned short* p = kbase + (size_t)(kc + t * 16) * CC + dk * 32;
                bk[t][dk].u[0] = *(const uint4*)(p);
                bk[t][dk].u[1] = *(const uint4*)(p + 16);
            }
        #pragma unroll
        for (int dt = 0; dt < 4; dt++) {
            const unsigned short* p = vbase + (size_t)(dt * 16) * TT + kc;
            bv[dt].u[0] = *(const uint4*)(p);
            bv[dt].u[1] = *(const uint4*)(p + 16);
        }
        if (kc + 32 <= q0) {               // prefetch next chunk (global_prefetch_b8)
            __builtin_prefetch(kbase + (size_t)(kc + 32) * CC);
            __builtin_prefetch(vbase + kc + 32);
        }

        // ---- S = (Q K^T) * scale, two 16-key tiles ----
        v8f s[2] = {};
        #pragma unroll
        for (int t = 0; t < 2; t++) {
            #pragma unroll
            for (int dk = 0; dk < 2; dk++)
                s[t] = __builtin_amdgcn_wmma_f32_16x16x32_bf16(
                    false, aq[dk].v, false, bk[t][dk].v, (short)0, s[t], false, false);
            s[t] = s[t] * ATT_SCALE;
        }
        // ---- causal mask (only needed near the diagonal) ----
        if (kc + 31 > q0) {
            #pragma unroll
            for (int t = 0; t < 2; t++)
                #pragma unroll
                for (int r = 0; r < 8; r++) {
                    int key = kc + t * 16 + lr;
                    int row = q0 + r + half * 8;
                    if (key > row) s[t][r] = -1e30f;
                }
        }
        // ---- online softmax update ----
        float alpha[8];
        #pragma unroll
        for (int r = 0; r < 8; r++) {
            float rm = rowmax16(fmaxf(s[0][r], s[1][r]));
            float mn = fmaxf(m[r], rm);
            alpha[r] = __expf(m[r] - mn);
            m[r] = mn;
            float p0 = __expf(s[0][r] - mn);
            float p1 = __expf(s[1][r] - mn);
            s[0][r] = p0; s[1][r] = p1;
            l[r] = l[r] * alpha[r] + rowsum16(p0 + p1);
        }
        #pragma unroll
        for (int dt = 0; dt < 4; dt++)
            #pragma unroll
            for (int r = 0; r < 8; r++) acc[dt][r] *= alpha[r];

        // ---- P: D-layout -> LDS -> A-layout (bf16) ----
        #pragma unroll
        for (int r = 0; r < 8; r++) {
            pLds[(r + half * 8) * 32 + 0  + lr] = f2bf(s[0][r]);
            pLds[(r + half * 8) * 32 + 16 + lr] = f2bf(s[1][r]);
        }
        asm volatile("s_wait_dscnt 0" ::: "memory");   // DS is in-order per wave
        BF16x16 pa;
        {
            const unsigned short* lp = pLds + lr * 32 + half * 8;
            pa.u[0] = *(const uint4*)(lp);
            pa.u[1] = *(const uint4*)(lp + 16);
        }
        // ---- acc += P V ----
        #pragma unroll
        for (int dt = 0; dt < 4; dt++)
            acc[dt] = __builtin_amdgcn_wmma_f32_16x16x32_bf16(
                false, pa.v, false, bv[dt].v, (short)0, acc[dt], false, false);
        asm volatile("" ::: "memory");   // keep LDS store/load ordering across iters
    }

    // ---- normalize and store bf16 ----
    #pragma unroll
    for (int dt = 0; dt < 4; dt++)
        #pragma unroll
        for (int r = 0; r < 8; r++) {
            int row = q0 + r + half * 8;
            int col = h * HSZ + dt * 16 + lr;
            out[(size_t)(b * TT + row) * CC + col] = f2bf(acc[dt][r] / l[r]);
        }
}

// ---------------------------------------------------------------------------
// Host launcher
// ---------------------------------------------------------------------------
extern "C" void kernel_launch(void* const* d_in, const int* in_sizes, int n_in,
                              void* d_out, int out_size, void* d_ws, size_t ws_size,
                              hipStream_t stream) {
    (void)in_sizes; (void)n_in; (void)out_size; (void)ws_size;

    const float* x    = (const float*)d_in[0];
    const float* Wq   = (const float*)d_in[1];
    const float* Wk   = (const float*)d_in[2];
    const float* Wv   = (const float*)d_in[3];
    const float* Wp   = (const float*)d_in[4];
    const float* bp   = (const float*)d_in[5];
    const float* W1   = (const float*)d_in[6];
    const float* b1   = (const float*)d_in[7];
    const float* W2   = (const float*)d_in[8];
    const float* b2   = (const float*)d_in[9];
    const float* g1   = (const float*)d_in[10];
    const float* bt1  = (const float*)d_in[11];
    const float* g2   = (const float*)d_in[12];
    const float* bt2  = (const float*)d_in[13];
    float* out        = (float*)d_out;

    char* w = (char*)d_ws;
    auto alloc = [&](size_t bytes) -> void* {
        void* p = (void*)w;
        w += (bytes + 255) & ~(size_t)255;
        return p;
    };
    unsigned short* WqT = (unsigned short*)alloc((size_t)CC * CC * 2);
    unsigned short* WkT = (unsigned short*)alloc((size_t)CC * CC * 2);
    unsigned short* WvT = (unsigned short*)alloc((size_t)CC * CC * 2);
    unsigned short* WpT = (unsigned short*)alloc((size_t)CC * CC * 2);
    unsigned short* W1T = (unsigned short*)alloc((size_t)CF * CC * 2);
    unsigned short* W2T = (unsigned short*)alloc((size_t)CC * CF * 2);
    unsigned short* hB  = (unsigned short*)alloc((size_t)MTOT * CC * 2);
    unsigned short* qB  = (unsigned short*)alloc((size_t)MTOT * CC * 2);
    unsigned short* kB  = (unsigned short*)alloc((size_t)MTOT * CC * 2);
    unsigned short* vTB = (unsigned short*)alloc((size_t)MTOT * CC * 2);
    unsigned short* atB = (unsigned short*)alloc((size_t)MTOT * CC * 2);
    float*          x2  = (float*)        alloc((size_t)MTOT * CC * 4);
    unsigned short* h2B = (unsigned short*)alloc((size_t)MTOT * CC * 2);
    unsigned short* ffB = (unsigned short*)alloc((size_t)MTOT * CF * 2);

    dim3 tb(32, 8);
    transpose_bf16<<<dim3(CC / 32, CC / 32), tb, 0, stream>>>(Wq, WqT, CC, CC);
    transpose_bf16<<<dim3(CC / 32, CC / 32), tb, 0, stream>>>(Wk, WkT, CC, CC);
    transpose_bf16<<<dim3(CC / 32, CC / 32), tb, 0, stream>>>(Wv, WvT, CC, CC);
    transpose_bf16<<<dim3(CC / 32, CC / 32), tb, 0, stream>>>(Wp, WpT, CC, CC);
    transpose_bf16<<<dim3(CF / 32, CC / 32), tb, 0, stream>>>(W1, W1T, CC, CF);
    transpose_bf16<<<dim3(CC / 32, CF / 32), tb, 0, stream>>>(W2, W2T, CF, CC);

    ln_bf16<<<MTOT, 256, 0, stream>>>(x, g1, bt1, hB);

    dim3 g1d(CC / 128, MTOT / 128);
    gemm_wmma<0><<<g1d, 256, 0, stream>>>(hB, WqT, nullptr, nullptr, nullptr, qB,  MTOT, CC, CC);
    gemm_wmma<0><<<g1d, 256, 0, stream>>>(hB, WkT, nullptr, nullptr, nullptr, kB,  MTOT, CC, CC);
    gemm_wmma<1><<<g1d, 256, 0, stream>>>(hB, WvT, nullptr, nullptr, nullptr, vTB, MTOT, CC, CC);

    attn_flash<<<BB * HH * (TT / 16), 32, 0, stream>>>(qB, kB, vTB, atB);

    gemm_wmma<2><<<g1d, 256, 0, stream>>>(atB, WpT, bp, x, x2, nullptr, MTOT, CC, CC);

    ln_bf16<<<MTOT, 256, 0, stream>>>(x2, g2, bt2, h2B);

    dim3 gff(CF / 128, MTOT / 128);
    gemm_wmma<3><<<gff, 256, 0, stream>>>(h2B, W1T, b1, nullptr, nullptr, ffB, MTOT, CF, CC);

    gemm_wmma<2><<<g1d, 256, 0, stream>>>(ffB, W2T, b2, x2, out, nullptr, MTOT, CC, CF);
}
